// RDL_SNNModel_44667659878462
// MI455X (gfx1250) — matrix operand order repaired and checked by reading, not compiled, not run on hardware
//
#include <hip/hip_runtime.h>
#include <hip/hip_bf16.h>

// SNN forward, single persistent kernel:
//   100 timesteps x (in->h, 3x h->h, h->out) Leaky layers.
// f16 WMMA (v_wmma_f32_16x16x32_f16) GEMMs, f32 accumulate, fused leaky/spike
// epilogue. B tiles staged into LDS via global_load_async_to_lds_b128 and
// shared by the 8 waves of a block; device-wide atomic grid barrier between
// layers replaces 500 kernel launches. Explicit address-space-qualified
// pointers force global_load / ds_load codegen (no flat_load).

#define USE_ASYNC_LDS 1

typedef __attribute__((ext_vector_type(16))) _Float16 v16h;
typedef __attribute__((ext_vector_type(8)))  _Float16 v8h;
typedef __attribute__((ext_vector_type(8)))  float    v8f;

// address-space-qualified element & vector types
typedef __attribute__((address_space(1))) float    gfloat;
typedef __attribute__((address_space(1))) _Float16 ghalf;
typedef __attribute__((address_space(3))) _Float16 lhalf;
typedef __attribute__((address_space(1))) v8f  gv8f;
typedef __attribute__((address_space(1))) v8h  gv8h;
typedef __attribute__((address_space(3))) v8h  lv8h;
typedef __attribute__((address_space(3))) v16h lv16h;

#define M_BATCH 512
#define N_HID 1024
#define N_IN  512
#define N_OUT 35
#define N_OUT_PAD 48
#define T_STEPS 100

#define NBLOCKS 128
#define NTHREADS 256          // 8 waves
#define CHUNK_K 128
#define STAGE_HALVES (32 * CHUNK_K)   // 32 cols x 128 k = 8KB per buffer

// ---------------- weight conversion kernels ----------------

__global__ void cvt_f32_f16_kernel(const float* __restrict__ in,
                                   _Float16* __restrict__ out, int n) {
    int i = blockIdx.x * blockDim.x + threadIdx.x;
    if (i < n) out[i] = (_Float16)in[i];
}

__global__ void cvt_wout_pad_kernel(const float* __restrict__ in,
                                    _Float16* __restrict__ out) {
    int i = blockIdx.x * blockDim.x + threadIdx.x;
    if (i >= N_OUT_PAD * N_HID) return;
    int row = i >> 10;
    int col = i & (N_HID - 1);
    out[i] = (row < N_OUT) ? (_Float16)in[row * N_HID + col] : (_Float16)0.0f;
}

// ---------------- LDS staging of B (weight) tiles ----------------
// LDS layout = exact per-lane WMMA B-fragment order: fragment f (=ks*2+s)
// occupies halves [f*512 .. f*512+511]; lane L's 32B slice at f*512 + L*16.
// Conflict-free consumption. 512 slots x 16B staged by 256 threads.

__device__ __forceinline__ void stage_tile(const ghalf* __restrict__ w,
                                           int ldw, int n0, int kc0,
                                           lhalf* dst) {
#pragma unroll
    for (int j = 0; j < 2; ++j) {
        const int slot = threadIdx.x + j * NTHREADS;   // 0..511
        const int f    = slot >> 6;                    // fragment 0..7
        const int ks   = f >> 1;
        const int s    = f & 1;
        const int lp   = (slot >> 1) & 31;             // consumer lane
        const int half = slot & 1;
        const int o    = n0 + s * 16 + (lp & 15);
        const int kb   = kc0 + ks * 32 + (lp >> 4) * 8 + half * 16;
        const ghalf* src = w + (size_t)o * ldw + kb;
        lhalf* d = dst + (size_t)slot * 8;
#if USE_ASYNC_LDS
        unsigned loff = (unsigned)(uintptr_t)d;        // AS3 ptr == LDS offset
        unsigned long long ga = (unsigned long long)(uintptr_t)src;
        asm volatile("global_load_async_to_lds_b128 %0, %1, off"
                     :: "v"(loff), "v"(ga) : "memory");
#else
        *(lv8h*)d = *(const gv8h*)src;
#endif
    }
}

__device__ __forceinline__ void stage_wait() {
#if USE_ASYNC_LDS
    asm volatile("s_wait_asynccnt 0x0" ::: "memory");
#endif
}

// ---------------- grid-wide barrier (persistent kernel) ----------------

__device__ __forceinline__ void grid_sync(unsigned* cnt, unsigned* gen) {
    __threadfence();          // every wave completes/flushes its own stores
    __syncthreads();
    if (threadIdx.x == 0) {
        unsigned g = __hip_atomic_load(gen, __ATOMIC_RELAXED,
                                       __HIP_MEMORY_SCOPE_AGENT);
        unsigned a = __hip_atomic_fetch_add(cnt, 1u, __ATOMIC_ACQ_REL,
                                            __HIP_MEMORY_SCOPE_AGENT);
        if (a == (unsigned)(NBLOCKS - 1)) {
            __hip_atomic_store(cnt, 0u, __ATOMIC_RELAXED,
                               __HIP_MEMORY_SCOPE_AGENT);
            __hip_atomic_fetch_add(gen, 1u, __ATOMIC_RELEASE,
                                   __HIP_MEMORY_SCOPE_AGENT);
        } else {
            while (__hip_atomic_load(gen, __ATOMIC_RELAXED,
                                     __HIP_MEMORY_SCOPE_AGENT) == g)
                __builtin_amdgcn_s_sleep(2);
        }
    }
    __syncthreads();
    __threadfence();          // acquire: invalidate stale lines
}

// ---------------- fused leaky/spike epilogue ----------------

template<int NT, bool FINAL>
__device__ __forceinline__ void leaky_epilogue(const v8f* acc, int bm, int n0,
        const gfloat* __restrict__ bias, const gfloat* __restrict__ beta,
        const gfloat* __restrict__ thr, gfloat* __restrict__ memst,
        ghalf* __restrict__ spk_h, gfloat* __restrict__ spk_f,
        int nvalid, int ldm) {
    const int lane  = threadIdx.x & 31;
    const int nlan  = lane & 15;
    const int mbase = bm * 16 + (lane >> 4) * 8;   // D rows: VGPR r -> mbase+r
#pragma unroll
    for (int s = 0; s < NT; ++s) {
        const int o = n0 + s * 16 + nlan;
        float bi, be, th;
        if constexpr (FINAL) {
            const bool ok = (o < nvalid);
            bi = ok ? bias[o] : 0.f;
            be = ok ? beta[o] : 0.f;
            th = ok ? thr[o]  : 1.f;
        } else { bi = bias[o]; be = beta[o]; th = thr[o]; }
        const float bc = fminf(fmaxf(be, 0.f), 1.f);
#pragma unroll
        for (int r = 0; r < 8; ++r) {
            const int brow = mbase + r;
            float mv = memst[(size_t)brow * ldm + o];
            const float reset = (mv - th > 0.f) ? th : 0.f;   // reset*thr
            mv = bc * mv + (acc[s][r] + bi) - reset;
            memst[(size_t)brow * ldm + o] = mv;
            const float sp = (mv - th > 0.f) ? 1.f : 0.f;
            if constexpr (FINAL) {
                if (o < nvalid) spk_f[(size_t)brow * nvalid + o] = sp;
            } else {
                spk_h[(size_t)brow * ldm + o] = (_Float16)sp;
            }
        }
    }
}

// ---------------- hidden / input layer: LDS-staged GEMM + leaky ----------

template<int K, bool AF32>
__device__ __forceinline__ void layer_hidden(const void* a_src_v,
        const ghalf* __restrict__ w,
        const gfloat* __restrict__ bias, const gfloat* __restrict__ beta,
        const gfloat* __restrict__ thr, gfloat* __restrict__ memst,
        ghalf* __restrict__ spk_out, lhalf* lds0, lhalf* lds1) {
    const int lane  = threadIdx.x & 31;
    const int wv    = threadIdx.x >> 5;          // wave in block 0..7
    const int strip = blockIdx.x >> 2;           // 0..31 (32-col N strip)
    const int bm    = (blockIdx.x & 3) * 8 + wv; // 0..31 batch tile
    const int n0    = strip * 32;
    const int m  = lane & 15, hs = lane >> 4;
    const int arow = bm * 16 + m;
    const ghalf*  ah = (const ghalf*)a_src_v;
    const gfloat* af = (const gfloat*)a_src_v;

    const v8f vz = {0.f, 0.f, 0.f, 0.f, 0.f, 0.f, 0.f, 0.f};
    v8f acc[2] = {vz, vz};
    constexpr int NC = K / CHUNK_K;

    stage_tile(w, K, n0, 0, lds0);
    stage_wait();
    __syncthreads();
    for (int kc = 0; kc < NC; ++kc) {
        const lhalf* cur = (kc & 1) ? lds1 : lds0;
        if (kc + 1 < NC)
            stage_tile(w, K, n0, (kc + 1) * CHUNK_K, (kc & 1) ? lds0 : lds1);
#pragma unroll
        for (int ks = 0; ks < 4; ++ks) {
            const int kb = kc * CHUNK_K + ks * 32 + hs * 8;
            v16h a;
            if constexpr (AF32) {
                v8f lo = *(const gv8f*)(af + (size_t)arow * K + kb);
                v8f hi = *(const gv8f*)(af + (size_t)arow * K + kb + 16);
#pragma unroll
                for (int i = 0; i < 8; ++i) {
                    a[i]     = (_Float16)lo[i];
                    a[i + 8] = (_Float16)hi[i];
                }
            } else {
                v8h lo = *(const gv8h*)(ah + (size_t)arow * K + kb);
                v8h hi = *(const gv8h*)(ah + (size_t)arow * K + kb + 16);
                a = __builtin_shufflevector(lo, hi, 0, 1, 2, 3, 4, 5, 6, 7,
                                            8, 9, 10, 11, 12, 13, 14, 15);
            }
#pragma unroll
            for (int s = 0; s < 2; ++s) {
                v16h bf = *(const lv16h*)(cur + ((size_t)(ks * 2 + s) * 32 + lane) * 16);
                acc[s] = __builtin_amdgcn_wmma_f32_16x16x32_f16(
                    false, a, false, bf, (short)0, acc[s], false, false);
            }
        }
        stage_wait();
        __syncthreads();
    }
    leaky_epilogue<2, false>(acc, bm, n0, bias, beta, thr, memst, spk_out,
                             nullptr, N_HID, N_HID);
}

// ---------------- readout layer (small: direct global B loads) -----------

__device__ __forceinline__ void layer_readout(const ghalf* __restrict__ a_src,
        const ghalf* __restrict__ w, const gfloat* __restrict__ bias,
        const gfloat* __restrict__ beta, const gfloat* __restrict__ thr,
        gfloat* __restrict__ memst, gfloat* __restrict__ spk_f) {
    if (blockIdx.x >= 4) return;
    const int lane = threadIdx.x & 31;
    const int wv   = threadIdx.x >> 5;
    const int bm   = blockIdx.x * 8 + wv;        // 0..31
    const int m = lane & 15, hs = lane >> 4;
    const int arow = bm * 16 + m;
    const int nlan = lane & 15;
    const v8f vz = {0.f, 0.f, 0.f, 0.f, 0.f, 0.f, 0.f, 0.f};
    v8f acc[3] = {vz, vz, vz};
    for (int k0 = 0; k0 < N_HID; k0 += 32) {
        const int kb = k0 + hs * 16;
        v8h lo = *(const gv8h*)(a_src + (size_t)arow * N_HID + kb);
        v8h hi = *(const gv8h*)(a_src + (size_t)arow * N_HID + kb + 8);
        v16h a = __builtin_shufflevector(lo, hi, 0, 1, 2, 3, 4, 5, 6, 7,
                                         8, 9, 10, 11, 12, 13, 14, 15);
        const int kbb = k0 + hs * 16;
#pragma unroll
        for (int s = 0; s < 3; ++s) {
            const int o = s * 16 + nlan;
            v16h bf = *(const __attribute__((address_space(1))) v16h*)
                          (w + (size_t)o * N_HID + kbb);
            acc[s] = __builtin_amdgcn_wmma_f32_16x16x32_f16(
                false, a, false, bf, (short)0, acc[s], false, false);
        }
    }
    leaky_epilogue<3, true>(acc, bm, 0, bias, beta, thr, memst, nullptr,
                            spk_f, N_OUT, N_OUT_PAD);
}

// ---------------- persistent kernel: full 100-step rollout ---------------

__global__ __launch_bounds__(NTHREADS)
void snn_persistent_kernel(const float* __restrict__ data_,
        const _Float16* __restrict__ Win_, const _Float16* __restrict__ Wh_,
        const _Float16* __restrict__ Wout_,
        const float* __restrict__ b_in_, const float* __restrict__ b_h_,
        const float* __restrict__ b_out_,
        const float* __restrict__ beta_h_, const float* __restrict__ thr_h_,
        const float* __restrict__ beta_o_, const float* __restrict__ thr_o_,
        float* __restrict__ mem_h_, float* __restrict__ mem_o_,
        _Float16* __restrict__ spkA_, _Float16* __restrict__ spkB_,
        float* __restrict__ out_, unsigned* __restrict__ sync) {
    __shared__ alignas(32) _Float16 ldsB[2][STAGE_HALVES];
    lhalf* lds0 = (lhalf*)&ldsB[0][0];
    lhalf* lds1 = (lhalf*)&ldsB[1][0];
    unsigned* cnt = sync;
    unsigned* gen = sync + 32;   // separate cacheline

    // force global address space for all device-memory operands
    const gfloat* data   = (const gfloat*)data_;
    const ghalf*  Win    = (const ghalf*)Win_;
    const ghalf*  Wh     = (const ghalf*)Wh_;
    const ghalf*  Wout   = (const ghalf*)Wout_;
    const gfloat* b_in   = (const gfloat*)b_in_;
    const gfloat* b_h    = (const gfloat*)b_h_;
    const gfloat* b_out  = (const gfloat*)b_out_;
    const gfloat* beta_h = (const gfloat*)beta_h_;
    const gfloat* thr_h  = (const gfloat*)thr_h_;
    const gfloat* beta_o = (const gfloat*)beta_o_;
    const gfloat* thr_o  = (const gfloat*)thr_o_;
    gfloat* mem_h = (gfloat*)mem_h_;
    gfloat* mem_o = (gfloat*)mem_o_;
    ghalf*  spkA  = (ghalf*)spkA_;
    ghalf*  spkB  = (ghalf*)spkB_;
    gfloat* out   = (gfloat*)out_;

    for (int t = 0; t < T_STEPS; ++t) {
        layer_hidden<N_IN, true>((const void*)(data + (size_t)t * M_BATCH * N_IN),
            Win, b_in, beta_h, thr_h, mem_h, spkA, lds0, lds1);
        grid_sync(cnt, gen);
        layer_hidden<N_HID, false>((const void*)spkA, Wh, b_h,
            beta_h + N_HID, thr_h + N_HID,
            mem_h + (size_t)M_BATCH * N_HID, spkB, lds0, lds1);
        grid_sync(cnt, gen);
        layer_hidden<N_HID, false>((const void*)spkB, Wh + (size_t)N_HID * N_HID,
            b_h + N_HID, beta_h + 2 * N_HID, thr_h + 2 * N_HID,
            mem_h + 2ull * M_BATCH * N_HID, spkA, lds0, lds1);
        grid_sync(cnt, gen);
        layer_hidden<N_HID, false>((const void*)spkA, Wh + 2ull * N_HID * N_HID,
            b_h + 2 * N_HID, beta_h + 3 * N_HID, thr_h + 3 * N_HID,
            mem_h + 3ull * M_BATCH * N_HID, spkB, lds0, lds1);
        grid_sync(cnt, gen);
        layer_readout(spkB, Wout, b_out, beta_o, thr_o, mem_o,
            out + (size_t)t * M_BATCH * N_OUT);
        grid_sync(cnt, gen);
    }
}

// ---------------- host launch ----------------

extern "C" void kernel_launch(void* const* d_in, const int* in_sizes, int n_in,
                              void* d_out, int out_size, void* d_ws, size_t ws_size,
                              hipStream_t stream) {
    const float* data   = (const float*)d_in[0];
    const float* W_in   = (const float*)d_in[1];
    const float* b_in   = (const float*)d_in[2];
    const float* W_h    = (const float*)d_in[3];
    const float* b_h    = (const float*)d_in[4];
    const float* W_out  = (const float*)d_in[5];
    const float* b_out  = (const float*)d_in[6];
    const float* beta_h = (const float*)d_in[7];
    const float* thr_h  = (const float*)d_in[8];
    const float* beta_o = (const float*)d_in[9];
    const float* thr_o  = (const float*)d_in[10];
    float* out = (float*)d_out;

    // workspace layout
    char* ws = (char*)d_ws;
    _Float16* Win_h  = (_Float16*)ws;                                   // 1 MB
    _Float16* Wh_h   = (_Float16*)(ws + (size_t)N_HID * N_IN * 2);      // 6 MB
    _Float16* Wout_h = (_Float16*)(ws + (size_t)N_HID * N_IN * 2
                                      + 3ull * N_HID * N_HID * 2);      // 96 KB
    char* mem_base   = (char*)Wout_h + (size_t)N_OUT_PAD * N_HID * 2;
    float* mem_h     = (float*)mem_base;                                 // 8 MB
    float* mem_o     = (float*)(mem_base + 4ull * M_BATCH * N_HID * 4);  // 96 KB
    unsigned* syncp  = (unsigned*)((char*)mem_o + (size_t)M_BATCH * N_OUT_PAD * 4);
    _Float16* spkA   = (_Float16*)((char*)syncp + 256);
    _Float16* spkB   = spkA + (size_t)M_BATCH * N_HID;

    // 1) weights -> f16 (deterministic each call)
    {
        int n = N_HID * N_IN;
        cvt_f32_f16_kernel<<<(n + 255) / 256, 256, 0, stream>>>(W_in, Win_h, n);
        n = 3 * N_HID * N_HID;
        cvt_f32_f16_kernel<<<(n + 255) / 256, 256, 0, stream>>>(W_h, Wh_h, n);
        n = N_OUT_PAD * N_HID;
        cvt_wout_pad_kernel<<<(n + 255) / 256, 256, 0, stream>>>(W_out, Wout_h);
    }

    // 2) zero membranes + barrier state (contiguous region)
    hipMemsetAsync(mem_h, 0,
                   4ull * M_BATCH * N_HID * 4 + (size_t)M_BATCH * N_OUT_PAD * 4
                   + 256, stream);

    // 3) one persistent kernel for the entire 100-step rollout
    snn_persistent_kernel<<<NBLOCKS, NTHREADS, 0, stream>>>(
        data, Win_h, Wh_h, Wout_h, b_in, b_h, b_out,
        beta_h, thr_h, beta_o, thr_o, mem_h, mem_o, spkA, spkB, out, syncp);
}
